// MultiScaleFrequencyEnhancement_60988535603863
// MI455X (gfx1250) — matrix operand order repaired and checked by reading, not compiled, not run on hardware
//
#include <hip/hip_runtime.h>

// MI455X / gfx1250, wave32. Multi-scale frequency enhancement reduced to
// per-patch circular convolutions expressed as f16 WMMA GEMMs.
//
// Key HW reasoning:
//  * per-patch rfft2->gain->irfft2 (ortho) == per-patch circular conv, i.e.
//    out = T_ps @ patch  -> batched 16x16x32 f16 WMMA GEMMs.
//  * workload is HBM-bound (~27 GFLOP vs 23.3 TB/s): intermediates stored f16
//    so outbuf+ybuf = 134 MB and fit the 192 MB global L2 (L2-resident
//    producer->consumer round trips); all accumulation stays f32.
//
// Passes:
//  1. init        : zero BN-stat accumulators
//  2. pool        : per-(b,c) spatial mean (compulsory 134 MB x read)
//  3. attn        : tiny MLP + softmax -> aw[B][3]
//  4. prep        : build h4/h8/h16 from gains, materialize T matrices (f16)
//  5. fuse_scales : WMMA GEMMs T_ps @ patches, weighted sum -> output (f16)
//  6. dwconv      : depthwise 3x3 + bias -> y (f16), per-channel sum/sumsq (f32)
//  7. bn_final    : finalize BN scale/shift
//  8. pointwise   : WMMA 64x64 GEMM: pw_w @ relu(BN(y)) + bias + x (f32 out)

typedef __attribute__((ext_vector_type(16))) _Float16 v16h;
typedef __attribute__((ext_vector_type(8)))  _Float16 v8h;
typedef __attribute__((ext_vector_type(8)))  float    v8f;

#define Bn 8
#define Cn 64
#define Hn 256
#define Wn 256
#define BN_EPS 1e-5f

// ---------------- WMMA helpers (contiguous k order, wave32) ----------------

__device__ __forceinline__ v8f wmma_f16(v16h a, v16h b, v8f c) {
  // D = A(16x32 f16) * B(32x16 f16) + C(16x16 f32)
  return __builtin_amdgcn_wmma_f32_16x16x32_f16(
      /*neg_a=*/false, a, /*neg_b=*/false, b,
      /*c_mod=*/(short)0, c, /*reuse_a=*/false, /*reuse_b=*/false);
}

// A-matrix 16x32 f16, k order used here (same as B): lanes 0-15 hold M=lane, K=0..15;
// lanes 16-31 hold M=lane-16, K=16..31. Row-major T => two 16B loads.
__device__ __forceinline__ v16h load_A_16x32(const _Float16* T, int ldk,
                                             int m0, int k0, int lane) {
  int M  = lane & 15;
  int kb = (lane >> 4) << 4;  // 0 or 16
  const _Float16* p = T + (size_t)(m0 + M) * ldk + k0 + kb;
  v8h lo = *(const v8h*)(p);
  v8h hi = *(const v8h*)(p + 8);
  v16h a;
#pragma unroll
  for (int i = 0; i < 8; ++i) { a[i] = lo[i]; a[i + 8] = hi[i]; }
  return a;
}

// B-matrix 32x16 f16, k order used here (same as A): lane n in 0..15 holds column n, K=0..15;
// lane n+16 holds column n, K=16..31. LDS staged [column][K] => one 32B load.
__device__ __forceinline__ v16h load_B_Kx16(const _Float16* Bcol, int ldk,
                                            int n0, int k0, int lane) {
  int n  = lane & 15;
  int kb = (lane >> 4) << 4;  // 0 or 16
  return *(const v16h*)(Bcol + (size_t)(n0 + n) * ldk + k0 + kb);
}

// ---------------- 1. init ----------------

__global__ void init_kernel(float* stats) {
  if (threadIdx.x < 128) stats[threadIdx.x] = 0.f;
}

// ---------------- 2. pooled mean ----------------

__global__ void __launch_bounds__(256) pool_kernel(const float* x, float* pooled) {
  int bc = blockIdx.x;  // b*Cn + c
  const float* p = x + (size_t)bc * (Hn * Wn);
  float s = 0.f;
  for (int i = threadIdx.x; i < Hn * Wn; i += 256) s += p[i];
  __shared__ float red[256];
  red[threadIdx.x] = s;
  __syncthreads();
  for (int off = 128; off > 0; off >>= 1) {
    if (threadIdx.x < off) red[threadIdx.x] += red[threadIdx.x + off];
    __syncthreads();
  }
  if (threadIdx.x == 0) pooled[bc] = red[0] * (1.f / (Hn * Wn));
}

// ---------------- 3. attention MLP + softmax ----------------

__global__ void attn_kernel(const float* pooled, const float* w1, const float* b1,
                            const float* w2, const float* b2, float* aw) {
  int b = threadIdx.x;
  if (b >= Bn) return;
  float h1[16];
  for (int j = 0; j < 16; ++j) {
    float s = b1[j];
    for (int c = 0; c < Cn; ++c) s += w1[j * Cn + c] * pooled[b * Cn + c];
    h1[j] = s > 0.f ? s : 0.f;
  }
  float lg[3];
  float mx = -1e30f;
  for (int i = 0; i < 3; ++i) {
    float s = b2[i];
    for (int j = 0; j < 16; ++j) s += w2[i * 16 + j] * h1[j];
    lg[i] = s;
    mx = s > mx ? s : mx;
  }
  float den = 0.f;
  for (int i = 0; i < 3; ++i) { lg[i] = __expf(lg[i] - mx); den += lg[i]; }
  for (int i = 0; i < 3; ++i) aw[b * 3 + i] = lg[i] / den;
}

// ---------------- 4. prep: h kernels, T matrices, f16 weights ----------------

// h[dy][dx] = (1/ps^2) * sum_{ky,kx} g[bin(ky,kx)] * cos(2*pi*(ky*dy+kx*dx)/ps)
__device__ float compute_h(int ps, const float* g, int dy, int dx) {
  const float rmax = 0.70710678f;  // sqrt(0.5), max radial freq on grid
  float s = 0.f;
  for (int ky = 0; ky < ps; ++ky) {
    float fy = ((ky <= ps / 2 - 1) ? (float)ky : (float)(ky - ps)) / (float)ps;
    for (int kx = 0; kx < ps; ++kx) {
      int kxr = (kx <= ps / 2) ? kx : (ps - kx);
      float fx = (float)kxr / (float)ps;
      float r = sqrtf(fy * fy + fx * fx) / (rmax + 1e-8f);
      int bin = (int)(r * 7.0f);
      bin = bin < 0 ? 0 : (bin > 7 ? 7 : bin);
      float ang = 6.28318530718f * (float)(ky * dy + kx * dx) / (float)ps;
      s += g[bin] * __cosf(ang);
    }
  }
  return s / (float)(ps * ps);
}

__global__ void __launch_bounds__(256)
prep_kernel(const float* g0, const float* g1, const float* g2, const float* pw_w,
            _Float16* T4p, _Float16* T8, _Float16* T16, _Float16* Wpw) {
  __shared__ float gs[24];
  __shared__ float h4[16], h8[64], h16[256];
  int tid = threadIdx.x;
  if (tid < 8) { gs[tid] = g0[tid]; gs[8 + tid] = g1[tid]; gs[16 + tid] = g2[tid]; }
  __syncthreads();
  // radial-gain circular-conv kernels
  h16[tid] = compute_h(16, gs + 16, tid >> 4, tid & 15);
  if (tid < 64) h8[tid] = compute_h(8, gs + 8, tid >> 3, tid & 7);
  if (tid < 16) h4[tid] = compute_h(4, gs, tid >> 2, tid & 3);
  __syncthreads();
  // T16[m][n] (256x256)
  for (int idx = tid; idx < 256 * 256; idx += 256) {
    int m = idx >> 8, n = idx & 255;
    int du = ((m >> 4) - (n >> 4)) & 15;
    int dv = ((m & 15) - (n & 15)) & 15;
    T16[idx] = (_Float16)h16[du * 16 + dv];
  }
  // T8[m][n] (64x64)
  for (int idx = tid; idx < 64 * 64; idx += 256) {
    int m = idx >> 6, n = idx & 63;
    int du = ((m >> 3) - (n >> 3)) & 7;
    int dv = ((m & 7) - (n & 7)) & 7;
    T8[idx] = (_Float16)h8[du * 8 + dv];
  }
  // T4 padded to K=32 (cols 16..31 zero) so the f16 WMMA K=32 shape applies
  for (int idx = tid; idx < 16 * 32; idx += 256) {
    int m = idx >> 5, n = idx & 31;
    float v = 0.f;
    if (n < 16) {
      int du = ((m >> 2) - (n >> 2)) & 3;
      int dv = ((m & 3) - (n & 3)) & 3;
      v = h4[du * 4 + dv];
    }
    T4p[idx] = (_Float16)v;
  }
  // pointwise weight to f16 (64x64, row = out channel)
  for (int idx = tid; idx < 64 * 64; idx += 256) Wpw[idx] = (_Float16)pw_w[idx];
}

// ---------------- 5. fused multiscale enhancement (WMMA) ----------------

__global__ void __launch_bounds__(128)
fuse_scales_kernel(const float* x, const float* aw, const _Float16* T4p,
                   const _Float16* T8, const _Float16* T16, _Float16* outbuf) {
  // workgroup = (b, c, 16-row band). 4 waves.
  __shared__ _Float16 B16[16 * 256];   // [patch16][k=256]        8 KB
  __shared__ _Float16 B8s[64 * 64];    // [patch8 ][k=64]         8 KB
  __shared__ _Float16 B4s[256 * 32];   // [patch4 ][k=32, padded] 16 KB
  __shared__ float    acc[16 * 256];   // f32 accumulator        16 KB

  int wg = blockIdx.x;
  int band = wg & 15;
  int c = (wg >> 4) & 63;
  int b = wg >> 10;
  const float* xp = x + ((((size_t)b * Cn + c) * Hn + band * 16)) * Wn;
  float a0 = aw[b * 3 + 0], a1 = aw[b * 3 + 1], a2 = aw[b * 3 + 2];
  int tid = threadIdx.x;

  // stage band (16x256 f32) into the three per-scale B layouts (f16)
  for (int p = tid; p < 4096; p += 128) {
    int u = p >> 8, col = p & 255;
    _Float16 v = (_Float16)xp[u * Wn + col];
    B16[(col >> 4) * 256 + u * 16 + (col & 15)] = v;
    B8s[((u >> 3) * 32 + (col >> 3)) * 64 + (u & 7) * 8 + (col & 7)] = v;
    B4s[((u >> 2) * 64 + (col >> 2)) * 32 + (u & 3) * 4 + (col & 3)] = v;
    acc[p] = 0.f;
  }
  for (int p = tid; p < 4096; p += 128)  // zero K-pad of ps4 operand
    B4s[(p >> 4) * 32 + 16 + (p & 15)] = (_Float16)0.f;
  __syncthreads();

  int wave = tid >> 5, lane = tid & 31;
  int nlo = lane & 15;
  int mhi = (lane < 16) ? 0 : 8;

  // --- ps = 16: out(256x16) = T16(256x256) @ B16 ; 16 m-tiles, K loop of 8
  for (int t = wave; t < 16; t += 4) {
    v8f d = {};
    for (int k0 = 0; k0 < 256; k0 += 32) {
      v16h A = load_A_16x32(T16, 256, t * 16, k0, lane);
      v16h Bm = load_B_Kx16(B16, 256, 0, k0, lane);
      d = wmma_f16(A, Bm, d);
    }
#pragma unroll
    for (int r = 0; r < 8; ++r) {
      int mr = r + mhi;                       // v within patch row t
      acc[t * 256 + nlo * 16 + mr] += a2 * d[r];
    }
  }
  __syncthreads();

  // --- ps = 8: two sub-bands, out(64x32) = T8 @ B8 ; 16 tiles total
  for (int q = wave; q < 16; q += 4) {
    int s = q >> 3, mt = (q >> 1) & 3, nt = q & 1;
    v8f d = {};
    for (int k0 = 0; k0 < 64; k0 += 32) {
      v16h A = load_A_16x32(T8, 64, mt * 16, k0, lane);
      v16h Bm = load_B_Kx16(B8s + s * 32 * 64, 64, nt * 16, k0, lane);
      d = wmma_f16(A, Bm, d);
    }
    int n = nt * 16 + nlo;
#pragma unroll
    for (int r = 0; r < 8; ++r) {
      int m = mt * 16 + r + mhi;
      acc[(s * 8 + (m >> 3)) * 256 + n * 8 + (m & 7)] += a1 * d[r];
    }
  }
  __syncthreads();

  // --- ps = 4: four sub-bands, out(16x64) = T4p @ B4 ; K padded to 32
  for (int q = wave; q < 16; q += 4) {
    int qb = q >> 2, nt = q & 3;
    v16h A = load_A_16x32(T4p, 32, 0, 0, lane);
    v16h Bm = load_B_Kx16(B4s + qb * 64 * 32, 32, nt * 16, 0, lane);
    v8f d = {};
    d = wmma_f16(A, Bm, d);
    int n = nt * 16 + nlo;
#pragma unroll
    for (int r = 0; r < 8; ++r) {
      int m = r + mhi;
      acc[(qb * 4 + (m >> 2)) * 256 + n * 4 + (m & 3)] += a0 * d[r];
    }
  }
  __syncthreads();

  // f16 store: halves intermediate traffic; outbuf stays L2-resident
  _Float16* op = outbuf + ((((size_t)b * Cn + c) * Hn + band * 16)) * Wn;
  for (int p = tid; p < 4096; p += 128) op[p] = (_Float16)acc[p];
}

// ---------------- 6. depthwise 3x3 + bias + BN stats ----------------

__global__ void __launch_bounds__(256)
dwconv_kernel(const _Float16* outbuf, const float* dw_w, const float* dw_b,
              _Float16* ybuf, float* stats) {
  int wg = blockIdx.x;
  int band = wg & 15;
  int c = (wg >> 4) & 63;
  int b = wg >> 10;
  const _Float16* ip = outbuf + (((size_t)b * Cn + c)) * Hn * Wn;
  _Float16* yp = ybuf + ((((size_t)b * Cn + c) * Hn + band * 16)) * Wn;
  float wv[9];
#pragma unroll
  for (int i = 0; i < 9; ++i) wv[i] = dw_w[c * 9 + i];
  float bias = dw_b[c];
  int wcol = threadIdx.x;
  float s = 0.f, s2 = 0.f;
  for (int r = 0; r < 16; ++r) {
    int h = band * 16 + r;
    float a = bias;
#pragma unroll
    for (int ky = -1; ky <= 1; ++ky) {
      int hh = h + ky;
      if (hh < 0 || hh > Hn - 1) continue;
#pragma unroll
      for (int kx = -1; kx <= 1; ++kx) {
        int ww = wcol + kx;
        if (ww < 0 || ww > Wn - 1) continue;
        a += wv[(ky + 1) * 3 + (kx + 1)] * (float)ip[hh * Wn + ww];
      }
    }
    yp[r * Wn + wcol] = (_Float16)a;     // consumed as f16 WMMA operand anyway
    s += a;                              // stats accumulated in f32
    s2 += a * a;
  }
  __shared__ float red[256], red2[256];
  red[wcol] = s;
  red2[wcol] = s2;
  __syncthreads();
  for (int off = 128; off > 0; off >>= 1) {
    if (wcol < off) { red[wcol] += red[wcol + off]; red2[wcol] += red2[wcol + off]; }
    __syncthreads();
  }
  if (wcol == 0) {
    atomicAdd(&stats[c], red[0]);
    atomicAdd(&stats[64 + c], red2[0]);
  }
}

// ---------------- 7. BN finalize ----------------

__global__ void bn_final_kernel(const float* stats, const float* bn_g,
                                const float* bn_b, float* bnsc, float* bnsh) {
  int c = threadIdx.x;
  if (c >= Cn) return;
  const float invN = 1.f / (float)(Bn * Hn * Wn);
  float mean = stats[c] * invN;
  float var = stats[64 + c] * invN - mean * mean;
  float sc = bn_g[c] * rsqrtf(var + BN_EPS);
  bnsc[c] = sc;
  bnsh[c] = bn_b[c] - mean * sc;
}

// ---------------- 8. pointwise 1x1 conv (WMMA) + residual ----------------

__global__ void __launch_bounds__(128)
pointwise_kernel(const _Float16* ybuf, const float* x, const _Float16* Wpw,
                 const float* bnsc, const float* bnsh, const float* pw_b,
                 float* out) {
  __shared__ _Float16 zB[256 * 64];  // [pixel][channel], 32 KB
  int wg = blockIdx.x;
  int h = wg & 255;
  int b = wg >> 8;
  const _Float16* ypl = ybuf + (size_t)b * Cn * Hn * Wn + (size_t)h * Wn;
  int tid = threadIdx.x;
  for (int i = tid; i < Cn * 256; i += 128) {
    int c = i >> 8;
    int w = i & 255;
    float v = (float)ypl[(size_t)c * (Hn * Wn) + w] * bnsc[c] + bnsh[c];
    v = v > 0.f ? v : 0.f;
    zB[w * 64 + c] = (_Float16)v;
  }
  __syncthreads();
  int wave = tid >> 5, lane = tid & 31;
  int nlo = lane & 15;
  int mhi = (lane < 16) ? 0 : 8;
  const float* xpl = x + (size_t)b * Cn * Hn * Wn + (size_t)h * Wn;
  float* opl = out + (size_t)b * Cn * Hn * Wn + (size_t)h * Wn;
  for (int t = wave; t < 64; t += 4) {   // 4 o-tiles x 16 pixel-tiles
    int mt = t >> 4, nt = t & 15;
    v8f d = {};
    for (int k0 = 0; k0 < 64; k0 += 32) {
      v16h A = load_A_16x32(Wpw, 64, mt * 16, k0, lane);
      v16h Bm = load_B_Kx16(zB, 64, nt * 16, k0, lane);
      d = wmma_f16(A, Bm, d);
    }
    int w = nt * 16 + nlo;
#pragma unroll
    for (int r = 0; r < 8; ++r) {
      int o = mt * 16 + r + mhi;
      opl[(size_t)o * (Hn * Wn) + w] = d[r] + pw_b[o] + xpl[(size_t)o * (Hn * Wn) + w];
    }
  }
}

// ---------------- launch ----------------

extern "C" void kernel_launch(void* const* d_in, const int* in_sizes, int n_in,
                              void* d_out, int out_size, void* d_ws, size_t ws_size,
                              hipStream_t stream) {
  const float* x       = (const float*)d_in[0];
  const float* gains0  = (const float*)d_in[1];
  const float* gains1  = (const float*)d_in[2];
  const float* gains2  = (const float*)d_in[3];
  const float* attn_w1 = (const float*)d_in[4];
  const float* attn_b1 = (const float*)d_in[5];
  const float* attn_w2 = (const float*)d_in[6];
  const float* attn_b2 = (const float*)d_in[7];
  const float* dw_w    = (const float*)d_in[8];
  const float* dw_b    = (const float*)d_in[9];
  const float* bn_g    = (const float*)d_in[10];
  const float* bn_b    = (const float*)d_in[11];
  const float* pw_w    = (const float*)d_in[12];
  const float* pw_b    = (const float*)d_in[13];

  float* ws = (float*)d_ws;
  float* pooled = ws;                         // 512
  float* aw     = ws + 512;                   // 32
  float* stats  = ws + 544;                   // 128 (sum, sumsq per channel)
  float* bnsc   = ws + 672;                   // 64
  float* bnsh   = ws + 736;                   // 64
  _Float16* Wpw = (_Float16*)(ws + 800);      // 4096 halfs
  _Float16* T4p = (_Float16*)(ws + 2848);     // 512 halfs (16x32, K-padded)
  _Float16* T8  = (_Float16*)(ws + 3104);     // 4096 halfs
  _Float16* T16 = (_Float16*)(ws + 5152);     // 65536 halfs
  // f16 intermediates: 67 MB each; together they fit the 192 MB global L2
  _Float16* outbuf = (_Float16*)(ws + 37920); // B*C*H*W halfs
  _Float16* ybuf   = outbuf + (size_t)Bn * Cn * Hn * Wn;

  (void)in_sizes; (void)n_in; (void)out_size; (void)ws_size;

  init_kernel<<<1, 128, 0, stream>>>(stats);
  pool_kernel<<<Bn * Cn, 256, 0, stream>>>(x, pooled);
  attn_kernel<<<1, 32, 0, stream>>>(pooled, attn_w1, attn_b1, attn_w2, attn_b2, aw);
  prep_kernel<<<1, 256, 0, stream>>>(gains0, gains1, gains2, pw_w, T4p, T8, T16, Wpw);
  fuse_scales_kernel<<<Bn * Cn * 16, 128, 0, stream>>>(x, aw, T4p, T8, T16, outbuf);
  dwconv_kernel<<<Bn * Cn * 16, 256, 0, stream>>>(outbuf, dw_w, dw_b, ybuf, stats);
  bn_final_kernel<<<1, 64, 0, stream>>>(stats, bn_g, bn_b, bnsc, bnsh);
  pointwise_kernel<<<Bn * Hn, 128, 0, stream>>>(ybuf, x, Wpw, bnsc, bnsh, pw_b,
                                                (float*)d_out);
}
